// GRUDecoder_6021544149435
// MI455X (gfx1250) — matrix operand order repaired and checked by reading, not compiled, not run on hardware
//
#include <hip/hip_runtime.h>
#include <math.h>

// ---------------------------------------------------------------------------
// GRU decoder step + Bahdanau attention + vocab softmax for MI455X (gfx1250).
// Heavy GEMMs use V_WMMA_F32_16X16X32_BF16 (f32 accumulate); attention score
// GEMM fuses tanh(...)·v into the epilogue to avoid a 64MB intermediate.
// f32 activation stagings use GLOBAL_LOAD_ASYNC_TO_LDS_B128 (ASYNCcnt path).
// ---------------------------------------------------------------------------

typedef __attribute__((ext_vector_type(4)))  float          f32x4;
typedef __attribute__((ext_vector_type(4)))  unsigned short u16x4;
typedef __attribute__((ext_vector_type(8)))  unsigned short u16x8;
typedef __attribute__((ext_vector_type(16))) unsigned short u16x16;
typedef __attribute__((ext_vector_type(16))) __bf16         v16bf;
typedef __attribute__((ext_vector_type(8)))  float          v8f;

#define NB 32
#define TK 1024
#define HD 512
#define ED 256
#define NV 50257
#define LDA 520   // LDS row stride (ushorts): 1040B -> 16B aligned, bank-spread

__device__ __forceinline__ unsigned short f2bf(float f) {
  unsigned u = __float_as_uint(f);
  u += 0x7FFFu + ((u >> 16) & 1u);           // round-to-nearest-even
  return (unsigned short)(u >> 16);
}

__device__ __forceinline__ v16bf mk_frag(u16x8 lo, u16x8 hi) {
  u16x16 r = __builtin_shufflevector(lo, hi, 0,1,2,3,4,5,6,7,8,9,10,11,12,13,14,15);
  return __builtin_bit_cast(v16bf, r);
}

// A fragment (16-bit, 16xK tile in LDS): lane<16 -> M=lane, K
// [kk..kk+15] contiguous; lane>=16 -> M=lane-16, K [kk+16..kk+31].
__device__ __forceinline__ v16bf a_frag_lds(const unsigned short* lds, int row_base,
                                            int kk, int lane) {
  int half = lane >> 4;
  int m    = row_base + (lane & 15);
  const unsigned short* p = lds + m * LDA + kk + half * 16;
  u16x8 lo = *(const u16x8*)p;
  u16x8 hi = *(const u16x8*)(p + 8);
  return mk_frag(lo, hi);
}

// Async global->LDS copy, 16B per lane (CDNA5 ASYNCcnt path; no VGPR bounce).
__device__ __forceinline__ void async_ld_b128(void* lds_dst, const void* gsrc) {
  unsigned lds_off = (unsigned)(uintptr_t)lds_dst;
  asm volatile("global_load_async_to_lds_b128 %0, %1, off"
               :: "v"(lds_off), "v"(gsrc) : "memory");
}
__device__ __forceinline__ void wait_async0() {
  asm volatile("s_wait_asynccnt 0x0" ::: "memory");
}

// ---------------------------------------------------------------------------
// Small prologue kernels (sub-0.1 GFLOP each)
// ---------------------------------------------------------------------------

// Wh_w (HxH f32) -> bf16 once; L2-resident reuse by 2048 workgroups.
__global__ __launch_bounds__(256) void k_cvt_wh(const float* __restrict__ wh,
                                                unsigned short* __restrict__ whb) {
  int idx = (blockIdx.x * 256 + threadIdx.x) * 4;   // 65536 threads * 4 = 512*512
  f32x4 f = *(const f32x4*)(wh + idx);
  u16x4 s; s.x = f2bf(f.x); s.y = f2bf(f.y); s.z = f2bf(f.z); s.w = f2bf(f.w);
  *(u16x4*)(whb + idx) = s;
}

// x = concat(c_t_1, emb[y]) @ xc_w^T + xc_b      [32,256]
__global__ __launch_bounds__(256) void k_x(const int* __restrict__ y,
                                           const float* __restrict__ emb,
                                           const float* __restrict__ c_prev,
                                           const float* __restrict__ xc_w,
                                           const float* __restrict__ xc_b,
                                           float* __restrict__ x) {
  __shared__ __align__(16) float cat[HD + ED];
  int b = blockIdx.x, tid = threadIdx.x;
  int yy = y[b];
  for (int i = tid; i < HD; i += 256) cat[i] = c_prev[b * HD + i];
  if (tid < ED) cat[HD + tid] = emb[(size_t)yy * ED + tid];
  __syncthreads();
  int e = tid;
  float acc = xc_b[e];
  const float* w = xc_w + (size_t)e * (HD + ED);
  for (int k = 0; k < HD + ED; ++k) acc = fmaf(cat[k], w[k], acc);
  x[b * ED + e] = acc;
}

// single-step GRU (torch r,z,n layout)  -> h_new, also s_t output
__global__ __launch_bounds__(256) void k_gru(const float* __restrict__ x,
                                             const float* __restrict__ s_prev,
                                             const float* __restrict__ w_ih,
                                             const float* __restrict__ w_hh,
                                             const float* __restrict__ b_ih,
                                             const float* __restrict__ b_hh,
                                             float* __restrict__ h_new,
                                             float* __restrict__ s_out) {
  __shared__ __align__(16) float xl[ED];
  __shared__ __align__(16) float hl[HD];
  int b = blockIdx.x, tid = threadIdx.x;
  if (tid < ED) xl[tid] = x[b * ED + tid];
  for (int i = tid; i < HD; i += 256) hl[i] = s_prev[b * HD + i];
  __syncthreads();
  for (int j = tid; j < HD; j += 256) {
    float gir = b_ih[j], giz = b_ih[HD + j], gin = b_ih[2 * HD + j];
    const float* wr = w_ih + (size_t)j * ED;
    const float* wz = w_ih + (size_t)(HD + j) * ED;
    const float* wn = w_ih + (size_t)(2 * HD + j) * ED;
    for (int k = 0; k < ED; ++k) {
      float xv = xl[k];
      gir = fmaf(xv, wr[k], gir); giz = fmaf(xv, wz[k], giz); gin = fmaf(xv, wn[k], gin);
    }
    float ghr = b_hh[j], ghz = b_hh[HD + j], ghn = b_hh[2 * HD + j];
    const float* ur = w_hh + (size_t)j * HD;
    const float* uz = w_hh + (size_t)(HD + j) * HD;
    const float* un = w_hh + (size_t)(2 * HD + j) * HD;
    for (int k = 0; k < HD; ++k) {
      float hv = hl[k];
      ghr = fmaf(hv, ur[k], ghr); ghz = fmaf(hv, uz[k], ghz); ghn = fmaf(hv, un[k], ghn);
    }
    float r = 1.f / (1.f + expf(-(gir + ghr)));
    float z = 1.f / (1.f + expf(-(giz + ghz)));
    float n = tanhf(gin + r * ghn);
    float h = (1.f - z) * n + z * hl[j];
    h_new[b * HD + j] = h;
    s_out[b * HD + j] = h;
  }
}

// dec_fea = h_new @ dp_w^T + dp_b
__global__ __launch_bounds__(256) void k_decfea(const float* __restrict__ h_new,
                                                const float* __restrict__ dp_w,
                                                const float* __restrict__ dp_b,
                                                float* __restrict__ dec) {
  __shared__ __align__(16) float hl[HD];
  int b = blockIdx.x, tid = threadIdx.x;
  for (int i = tid; i < HD; i += 256) hl[i] = h_new[b * HD + i];
  __syncthreads();
  for (int j = tid; j < HD; j += 256) {
    float acc = dp_b[j];
    const float* w = dp_w + (size_t)j * HD;
    for (int k = 0; k < HD; ++k) acc = fmaf(hl[k], w[k], acc);
    dec[b * HD + j] = acc;
  }
}

// ---------------------------------------------------------------------------
// Attention scores: scores[b,t] = sum_h tanh( (enc@Wh^T)[b,t,h] + dec[b,h] )*v[h]
// 17.2 GFLOP GEMM via v_wmma_f32_16x16x32_bf16, fused epilogue (no [B,TK,H]
// intermediate). One WG per 16-row (t) tile; A tile staged bf16 in LDS; 8
// waves each own 4 of the 32 column (h) tiles.
// ---------------------------------------------------------------------------
__global__ __launch_bounds__(256) void k_attn_scores(const float* __restrict__ enc,
                                                     const unsigned short* __restrict__ whb,
                                                     const float* __restrict__ dec,
                                                     const float* __restrict__ v_w,
                                                     float* __restrict__ scores) {
  __shared__ __align__(16) unsigned short Albs[16 * LDA];
  __shared__ float ssc[16];
  int tid = threadIdx.x;
  int bt = blockIdx.x;               // 0..2047
  int b  = bt >> 6;
  int t0 = (bt & 63) << 4;
  // stage 16x512 f32 rows (contiguous 32KB) -> bf16 LDS (convert in-register)
  const float* arow = enc + ((size_t)(b * TK + t0)) * HD;
  for (int e = tid * 4; e < 16 * HD; e += 1024) {
    f32x4 f = *(const f32x4*)(arow + e);
    u16x4 s; s.x = f2bf(f.x); s.y = f2bf(f.y); s.z = f2bf(f.z); s.w = f2bf(f.w);
    int row = e >> 9, k = e & 511;
    *(u16x4*)(Albs + row * LDA + k) = s;
  }
  if (tid < 16) ssc[tid] = 0.f;
  __syncthreads();

  int lane = tid & 31, wave = tid >> 5;
  int half = lane >> 4, l16 = lane & 15;
  float sp[8] = {0.f, 0.f, 0.f, 0.f, 0.f, 0.f, 0.f, 0.f};

  for (int nt = wave; nt < HD / 16; nt += 8) {
    v8f c = {};
    int n_g = nt * 16 + l16;
    const unsigned short* brow = whb + (size_t)n_g * HD;
    for (int kk = 0; kk < HD; kk += 32) {
      v16bf a = a_frag_lds(Albs, 0, kk, lane);
      const unsigned short* bp = brow + kk + half * 16;  // B: 16 contiguous K/lane
      v16bf bb = mk_frag(*(const u16x8*)bp, *(const u16x8*)(bp + 8));
      c = __builtin_amdgcn_wmma_f32_16x16x32_bf16(false, a, false, bb,
                                                  (short)0, c, false, false);
    }
    float d  = dec[b * HD + n_g];
    float vv = v_w[n_g];
#pragma unroll
    for (int r = 0; r < 8; ++r) sp[r] += tanhf(c[r] + d) * vv;
  }
  // reduce across the 16 columns held by each lane half
#pragma unroll
  for (int r = 0; r < 8; ++r) {
    float s = sp[r];
    s += __shfl_xor(s, 1); s += __shfl_xor(s, 2);
    s += __shfl_xor(s, 4); s += __shfl_xor(s, 8);
    sp[r] = s;
  }
  if (l16 == 0) {
    int mb = half * 8;
#pragma unroll
    for (int r = 0; r < 8; ++r) atomicAdd(&ssc[mb + r], sp[r]);  // ds_add_f32
  }
  __syncthreads();
  if (tid < 16) scores[b * TK + t0 + tid] = ssc[tid];
}

// softmax over TK -> attn
__global__ __launch_bounds__(256) void k_softmax_tk(const float* __restrict__ scores,
                                                    float* __restrict__ attn) {
  __shared__ float red[256];
  int b = blockIdx.x, tid = threadIdx.x;
  const float* row = scores + b * TK;
  float m = -1e30f;
  for (int i = tid; i < TK; i += 256) m = fmaxf(m, row[i]);
  red[tid] = m; __syncthreads();
  for (int s = 128; s > 0; s >>= 1) { if (tid < s) red[tid] = fmaxf(red[tid], red[tid + s]); __syncthreads(); }
  float mx = red[0]; __syncthreads();
  float sum = 0.f;
  for (int i = tid; i < TK; i += 256) { float e = expf(row[i] - mx); attn[b * TK + i] = e; sum += e; }
  red[tid] = sum; __syncthreads();
  for (int s = 128; s > 0; s >>= 1) { if (tid < s) red[tid] += red[tid + s]; __syncthreads(); }
  float inv = 1.f / red[0]; __syncthreads();
  for (int i = tid; i < TK; i += 256) attn[b * TK + i] *= inv;
}

// c_t = attn @ encoder_outputs   (enc re-read is L2-resident).
// attn row staged memory->LDS with async B128 copies (no VGPR bounce).
__global__ __launch_bounds__(256) void k_context(const float* __restrict__ attn,
                                                 const float* __restrict__ enc,
                                                 float* __restrict__ ct_ws,
                                                 float* __restrict__ ct_out) {
  __shared__ __align__(16) float al[TK];
  int b = blockIdx.x, tid = threadIdx.x;
  async_ld_b128(&al[tid * 4], attn + b * TK + tid * 4);   // 256 lanes x 16B = 4KB
  wait_async0();
  __syncthreads();
  int h0 = tid, h1 = tid + 256;
  float a0 = 0.f, a1 = 0.f;
  for (int t = 0; t < TK; ++t) {
    float a = al[t];
    const float* er = enc + ((size_t)(b * TK + t)) * HD;
    a0 = fmaf(a, er[h0], a0);
    a1 = fmaf(a, er[h1], a1);
  }
  ct_ws[b * HD + h0] = a0; ct_ws[b * HD + h1] = a1;
  ct_out[b * HD + h0] = a0; ct_out[b * HD + h1] = a1;
}

// output = concat(h_new, c_t) @ out1_w^T + out1_b  (concat staged async)
__global__ __launch_bounds__(256) void k_out1(const float* __restrict__ h_new,
                                              const float* __restrict__ c_t,
                                              const float* __restrict__ out1_w,
                                              const float* __restrict__ out1_b,
                                              float* __restrict__ outv) {
  __shared__ __align__(16) float cat[2 * HD];
  int b = blockIdx.x, tid = threadIdx.x;
  const float* src = (tid < 128) ? (h_new + b * HD + tid * 4)
                                 : (c_t + b * HD + tid * 4 - HD);
  async_ld_b128(&cat[tid * 4], src);                      // 256 lanes x 16B = 4KB
  wait_async0();
  __syncthreads();
  for (int j = tid; j < HD; j += 256) {
    float acc = out1_b[j];
    const float* w = out1_w + (size_t)j * (2 * HD);
    for (int k = 0; k < 2 * HD; ++k) acc = fmaf(cat[k], w[k], acc);
    outv[b * HD + j] = acc;
  }
}

// ---------------------------------------------------------------------------
// logits = output @ out2_w^T + out2_b  (103MB weight-bound; f32 weights
// converted to bf16 in-register, fed to v_wmma_f32_16x16x32_bf16)
// ---------------------------------------------------------------------------
__global__ __launch_bounds__(256) void k_logits(const float* __restrict__ outv,
                                                const float* __restrict__ w2,
                                                const float* __restrict__ b2,
                                                float* __restrict__ logits) {
  __shared__ __align__(16) unsigned short Albs[32 * LDA];
  int tid = threadIdx.x;
  for (int e = tid * 4; e < 32 * HD; e += 1024) {
    f32x4 f = *(const f32x4*)(outv + e);
    u16x4 s; s.x = f2bf(f.x); s.y = f2bf(f.y); s.z = f2bf(f.z); s.w = f2bf(f.w);
    int row = e >> 9, k = e & 511;
    *(u16x4*)(Albs + row * LDA + k) = s;
  }
  __syncthreads();

  int lane = tid & 31, wave = tid >> 5;
  int half = lane >> 4, l16 = lane & 15;
  int nt = blockIdx.x * 8 + wave;
  if (nt * 16 >= NV) return;                 // only last block; after barrier

  int n_g = nt * 16 + l16;
  int n_ld = (n_g < NV) ? n_g : (NV - 1);    // clamp loads on the ragged tile
  const float* wrow = w2 + (size_t)n_ld * HD;

  v8f c0 = {}, c1 = {};
  for (int kk = 0; kk < HD; kk += 32) {
    v16bf a0 = a_frag_lds(Albs, 0,  kk, lane);
    v16bf a1 = a_frag_lds(Albs, 16, kk, lane);
    const f32x4* p = (const f32x4*)(wrow + kk + half * 16);
    f32x4 f0 = p[0], f1 = p[1], f2 = p[2], f3 = p[3];
    u16x16 bu;
    bu[0]  = f2bf(f0.x); bu[1]  = f2bf(f0.y); bu[2]  = f2bf(f0.z); bu[3]  = f2bf(f0.w);
    bu[4]  = f2bf(f1.x); bu[5]  = f2bf(f1.y); bu[6]  = f2bf(f1.z); bu[7]  = f2bf(f1.w);
    bu[8]  = f2bf(f2.x); bu[9]  = f2bf(f2.y); bu[10] = f2bf(f2.z); bu[11] = f2bf(f2.w);
    bu[12] = f2bf(f3.x); bu[13] = f2bf(f3.y); bu[14] = f2bf(f3.z); bu[15] = f2bf(f3.w);
    v16bf bb = __builtin_bit_cast(v16bf, bu);
    c0 = __builtin_amdgcn_wmma_f32_16x16x32_bf16(false, a0, false, bb, (short)0, c0, false, false);
    c1 = __builtin_amdgcn_wmma_f32_16x16x32_bf16(false, a1, false, bb, (short)0, c1, false, false);
  }
  if (n_g < NV) {
    float bias = b2[n_g];
#pragma unroll
    for (int r = 0; r < 8; ++r) {
      int m = r + half * 8;
      logits[(size_t)m * NV + n_g]        = c0[r] + bias;
      logits[(size_t)(m + 16) * NV + n_g] = c1[r] + bias;
    }
  }
}

// softmax over V -> final_dist
__global__ __launch_bounds__(256) void k_softmax_v(const float* __restrict__ logits,
                                                   float* __restrict__ fd) {
  __shared__ float red[256];
  int b = blockIdx.x, tid = threadIdx.x;
  const float* row = logits + (size_t)b * NV;
  float m = -1e30f;
  for (int i = tid; i < NV; i += 256) m = fmaxf(m, row[i]);
  red[tid] = m; __syncthreads();
  for (int s = 128; s > 0; s >>= 1) { if (tid < s) red[tid] = fmaxf(red[tid], red[tid + s]); __syncthreads(); }
  float mx = red[0]; __syncthreads();
  float sum = 0.f;
  for (int i = tid; i < NV; i += 256) { float e = expf(row[i] - mx); fd[(size_t)b * NV + i] = e; sum += e; }
  red[tid] = sum; __syncthreads();
  for (int s = 128; s > 0; s >>= 1) { if (tid < s) red[tid] += red[tid + s]; __syncthreads(); }
  float inv = 1.f / red[0]; __syncthreads();
  for (int i = tid; i < NV; i += 256) fd[(size_t)b * NV + i] *= inv;
}

// coverage passthrough (output buffer is poisoned, must be rewritten)
__global__ __launch_bounds__(256) void k_copy_cov(const float* __restrict__ cov,
                                                  float* __restrict__ out) {
  int i = blockIdx.x * 1024 + threadIdx.x;
  for (int j = 0; j < 4; ++j) out[i + j * 256] = cov[i + j * 256];
}

// ---------------------------------------------------------------------------
extern "C" void kernel_launch(void* const* d_in, const int* in_sizes, int n_in,
                              void* d_out, int out_size, void* d_ws, size_t ws_size,
                              hipStream_t stream) {
  (void)in_sizes; (void)n_in; (void)out_size; (void)ws_size;
  const int*   y      = (const int*)d_in[0];
  const float* s_t_1  = (const float*)d_in[2];
  const float* enc    = (const float*)d_in[3];
  const float* c_prev = (const float*)d_in[4];
  const float* cov    = (const float*)d_in[5];
  const float* emb    = (const float*)d_in[7];
  const float* xc_w   = (const float*)d_in[8];
  const float* xc_b   = (const float*)d_in[9];
  const float* wh_w   = (const float*)d_in[10];
  const float* w_ih   = (const float*)d_in[11];
  const float* w_hh   = (const float*)d_in[12];
  const float* b_ih   = (const float*)d_in[13];
  const float* b_hh   = (const float*)d_in[14];
  const float* dp_w   = (const float*)d_in[15];
  const float* dp_b   = (const float*)d_in[16];
  const float* v_w    = (const float*)d_in[17];
  const float* o1_w   = (const float*)d_in[18];
  const float* o1_b   = (const float*)d_in[19];
  const float* o2_w   = (const float*)d_in[20];
  const float* o2_b   = (const float*)d_in[21];

  // workspace layout (float offsets unless noted)
  float* ws_x   = (float*)d_ws;                                   // 32*256   @0
  float* ws_h   = ws_x + 8192;                                    // 32*512   @8192
  float* ws_df  = ws_x + 24576;                                   // 32*512   @24576
  float* ws_sc  = ws_x + 40960;                                   // 32*1024  @40960
  float* ws_o1  = ws_x + 73728;                                   // 32*512   @73728
  float* ws_ct  = ws_x + 90112;                                   // 32*512   @90112
  unsigned short* ws_whb = (unsigned short*)((char*)d_ws + 425984);   // 512*512 bf16
  float* ws_log = (float*)((char*)d_ws + 950272);                 // 32*50257

  // d_out: final_dist | s_t | c_t | attn | coverage
  float* o_fd  = (float*)d_out;
  float* o_st  = o_fd + (size_t)NB * NV;       // 1,608,224
  float* o_ct  = o_st + NB * HD;
  float* o_at  = o_ct + NB * HD;
  float* o_cov = o_at + NB * TK;

  k_cvt_wh      <<<256, 256, 0, stream>>>(wh_w, ws_whb);
  k_x           <<<NB, 256, 0, stream>>>(y, emb, c_prev, xc_w, xc_b, ws_x);
  k_gru         <<<NB, 256, 0, stream>>>(ws_x, s_t_1, w_ih, w_hh, b_ih, b_hh, ws_h, o_st);
  k_decfea      <<<NB, 256, 0, stream>>>(ws_h, dp_w, dp_b, ws_df);
  k_attn_scores <<<NB * TK / 16, 256, 0, stream>>>(enc, ws_whb, ws_df, v_w, ws_sc);
  k_softmax_tk  <<<NB, 256, 0, stream>>>(ws_sc, o_at);
  k_context     <<<NB, 256, 0, stream>>>(o_at, enc, ws_ct, o_ct);
  k_out1        <<<NB, 256, 0, stream>>>(ws_h, ws_ct, o1_w, o1_b, ws_o1);
  k_logits      <<<(((NV + 15) / 16) + 7) / 8, 256, 0, stream>>>(ws_o1, o2_w, o2_b, ws_log);
  k_softmax_v   <<<NB, 256, 0, stream>>>(ws_log, o_fd);
  k_copy_cov    <<<NB, 256, 0, stream>>>(cov, o_cov);
}
